// ParallelTacotron2Loss_80015240724973
// MI455X (gfx1250) — compile-verified, matrix operand order checked by reading, not run on hardware
//
#include <hip/hip_runtime.h>
#include <math.h>

typedef __attribute__((ext_vector_type(16))) _Float16 v16h;
typedef __attribute__((ext_vector_type(8)))  float    v8f;
typedef int v4i_ __attribute__((vector_size(16)));
typedef __attribute__((address_space(1))) v4i_* as1_v4i;
typedef __attribute__((address_space(3))) v4i_* as3_v4i;

#define B_  4
#define T_  512
#define C_  80
#define BIGF  100000000.0f
#define WARPF 1.0f
// softmin_gamma with gamma=0.1 via exp2/log2:
//   exp(-(x-m)/g) = exp2((m-x) * (1/g)*log2(e))
#define SM_SCALE 14.4269504088896340736f  // (1/0.1) * log2(e)
#define SM_GLOG  0.06931471805599453094f  // 0.1 * ln(2)

#if defined(__HIP_DEVICE_COMPILE__) && \
    __has_builtin(__builtin_amdgcn_global_load_async_to_lds_b128) && \
    __has_builtin(__builtin_amdgcn_s_wait_asynccnt)
#define USE_ASYNC_LDS 1
#else
#define USE_ASYNC_LDS 0
#endif

#if defined(__HIP_DEVICE_COMPILE__) && __has_builtin(__builtin_amdgcn_exp2f)
#define EXP2F(x) __builtin_amdgcn_exp2f(x)
#else
#define EXP2F(x) exp2f(x)
#endif

// |x| via v_max_num_f32 with source-neg modifier. Inline asm is opaque to the
// SLP vectorizer, which otherwise fuses the fabs chain into packed v2f32
// abs patterns that crash this clang-23-rc gfx1250 ISel.
__device__ __forceinline__ float abs_scalar(float x) {
  float r;
  asm("v_max_num_f32_e64 %0, %1, -%1" : "=v"(r) : "v"(x));
  return r;
}

// ---------------------------------------------------------------------------
// Kernel 1: D[b,i,j] = sum_c |pred[b,i,c] - targ[b,j,c]|
// Tile: 8 i-rows x 16 j-cols per 256-thread block (8 waves; wave w owns row
// i0+w and 16 cells j0..j0+15). Channel reduction is done on the matrix pipe:
// A = 16x32 f16 tile of |diff| (documented 16-bit A striping), B = all-ones,
// so every column of D = row-sums. 3 chained WMMAs cover C=80 (padded to 96).
// ---------------------------------------------------------------------------
__global__ __launch_bounds__(256) void pt2_costmat_wmma(
    const float* __restrict__ mel_pred, const float* __restrict__ mel_target,
    float* __restrict__ D) {
  __shared__ __align__(16) float predS[8 * C_];    // 8 rows  x 80 ch
  __shared__ __align__(16) float targS[16 * C_];   // 16 rows x 80 ch
  const int b  = blockIdx.z;
  const int i0 = blockIdx.y * 8;
  const int j0 = blockIdx.x * 16;
  const int t  = threadIdx.x;

  // Rows are contiguous in memory: stage two contiguous blocks (640 + 1280 f32).
  const float4* gpred = (const float4*)(mel_pred   + ((size_t)b * T_ + i0) * C_);
  const float4* gtarg = (const float4*)(mel_target + ((size_t)b * T_ + j0) * C_);
  float4* spred = (float4*)predS;
  float4* starg = (float4*)targS;
  const int NPRED4 = 8 * C_ / 4;    // 160 float4
  const int NTOT4  = 24 * C_ / 4;   // 480 float4

  for (int idx = t; idx < NTOT4; idx += 256) {
    const float4* src = (idx < NPRED4) ? (gpred + idx) : (gtarg + (idx - NPRED4));
    float4*       dst = (idx < NPRED4) ? (spred + idx) : (starg + (idx - NPRED4));
#if USE_ASYNC_LDS
    __builtin_amdgcn_global_load_async_to_lds_b128((as1_v4i)src, (as3_v4i)dst, 0, 0);
#else
    *dst = *src;
#endif
  }
#if USE_ASYNC_LDS
  __builtin_amdgcn_s_wait_asynccnt(0);
#endif
  __syncthreads();

  const int lane = t & 31;
  const int w    = t >> 5;              // wave id -> i row
  const int m    = lane & 15;           // cell index (j = j0+m); A-matrix row M
  const int hi8  = (lane & 16) ? 8 : 0; // 16-bit A striping: upper half-wave K+8

  const float* prow = predS + w * C_;
  const float* trow = targS + m * C_;

  v16h ones;
#pragma unroll
  for (int e = 0; e < 16; ++e) ones[e] = (_Float16)1.0f;

  v8f acc = {};
#pragma unroll
  for (int kb = 0; kb < 96; kb += 32) {
    v16h a;
#pragma unroll
    for (int r = 0; r < 8; ++r) {       // A VGPR index
#pragma unroll
      for (int h = 0; h < 2; ++h) {     // low/high f16 half
        // 16-bit 16x32 A layout: r<4 -> K = hi8 + 2r + h ; r>=4 -> K = 16 + hi8 + 2(r-4) + h
        const int k = kb + ((r >> 2) << 4) + hi8 + ((r & 3) << 1) + h;
        float ad = 0.0f;
        if (k < C_) ad = abs_scalar(prow[k] - trow[k]);
        a[2 * r + h] = (_Float16)ad;
      }
    }
    // D = A x ones + acc : every column holds the per-cell channel sum
    acc = __builtin_amdgcn_wmma_f32_16x16x32_f16(false, a, false, ones,
                                                 (short)0, acc, false, false);
  }

  // C/D layout: vgpr v, lanes 0-15 -> M=v, lanes 16-31 -> M=v+8 (any column N).
  float* Drow = D + (((size_t)b * T_) + (size_t)(i0 + w)) * T_ + j0;
  if (lane < 8)                      Drow[lane]            = acc[lane];
  else if (lane >= 16 && lane < 24)  Drow[8 + (lane - 16)] = acc[lane - 16];
}

// ---------------------------------------------------------------------------
// Kernel 2: soft-DTW anti-diagonal wavefront. One 512-thread block per batch.
// Three rotating LDS diagonals of R (indexed by row p, p+q = s):
//   C[i+1] = D[i, k-i] + softmin(A[i], B[i]+warp, B[i+1]+warp)
// Exactly one barrier per diagonal (writes go to the third buffer).
// ---------------------------------------------------------------------------
__global__ __launch_bounds__(512) void pt2_sdtw_wavefront(
    const float* __restrict__ D, float* __restrict__ sdtw) {
  const int N = T_, M = T_;
  const int b = blockIdx.x;
  const int t = threadIdx.x;            // row index i; writes p = t+1
  __shared__ float diag[3][T_ + 4];

  for (int p = t; p < T_ + 4; p += 512) {
    diag[0][p] = BIGF; diag[1][p] = BIGF; diag[2][p] = BIGF;
  }
  if (t == 0) diag[0][0] = 0.0f;        // R[0][0]
  __syncthreads();

  const float* Db = D + (size_t)b * N * M;
  int ia = 0, ib = 1, ic = 2;

  for (int k = 0; k < N + M - 1; ++k) {
    const int j = k - t;
    float cv = BIGF;
    if (0 <= j && j < M) {
      const float a  = diag[ia][t];             // R[i][j]
      const float up = diag[ib][t] + WARPF;     // R[i][j+1] + warp
      const float lf = diag[ib][t + 1] + WARPF; // R[i+1][j] + warp
      const float mn = fminf(a, fminf(up, lf));
      const float s  = EXP2F((mn - a)  * SM_SCALE)
                     + EXP2F((mn - up) * SM_SCALE)
                     + EXP2F((mn - lf) * SM_SCALE);
      const float smin = mn - SM_GLOG * __log2f(s);
      cv = Db[t * M + j] + smin;
    }
    diag[ic][t + 1] = cv;                // distinct buffer: no read/write overlap
    if (t == 0) diag[ic][0] = BIGF;      // R[0][q>0] stays BIG
    __syncthreads();
    const int tmp = ia; ia = ib; ib = ic; ic = tmp;
  }

  if (t == 0) sdtw[b] = diag[ib][N];     // R[N][M]
}

// ---------------------------------------------------------------------------
// Kernel 3: duration loss (512 elems) + mel loss -> scalar
// ---------------------------------------------------------------------------
__global__ __launch_bounds__(512) void pt2_finalize(
    const float* __restrict__ log_d_pred, const int* __restrict__ d_target,
    const float* __restrict__ sdtw, float* __restrict__ out) {
  __shared__ float red[512];
  const int t = threadIdx.x;
  const float ld = __logf((float)d_target[t] + 1.0f);
  red[t] = abs_scalar(log_d_pred[t] - ld);
  __syncthreads();
  for (int s = 256; s > 0; s >>= 1) {
    if (t < s) red[t] += red[t + s];
    __syncthreads();
  }
  if (t == 0) {
    const float dur = red[0] * (1.0f / 512.0f);
    float mel = 0.0f;
    for (int bb = 0; bb < B_; ++bb) mel += sdtw[bb];
    mel *= (1.0f / ((float)B_ * (float)T_));
    out[0] = mel + dur;
  }
}

extern "C" void kernel_launch(void* const* d_in, const int* in_sizes, int n_in,
                              void* d_out, int out_size, void* d_ws, size_t ws_size,
                              hipStream_t stream) {
  const float* mel_pred   = (const float*)d_in[0];   // [4,512,80] f32
  const float* mel_target = (const float*)d_in[1];   // [4,512,80] f32
  const float* log_d_pred = (const float*)d_in[2];   // [4,128]    f32
  const int*   d_target   = (const int*)d_in[3];     // [4,128]    i32

  float* D    = (float*)d_ws;                        // 4*512*512 f32 = 4 MB
  float* sdtw = D + (size_t)B_ * T_ * T_;            // +4 f32

  dim3 gA(T_ / 16, T_ / 8, B_);                      // (32, 64, 4)
  pt2_costmat_wmma<<<gA, 256, 0, stream>>>(mel_pred, mel_target, D);
  pt2_sdtw_wavefront<<<B_, 512, 0, stream>>>(D, sdtw);
  pt2_finalize<<<1, 512, 0, stream>>>(log_d_pred, d_target, sdtw, (float*)d_out);
}
